// Attention_89584427860152
// MI455X (gfx1250) — matrix operand order repaired and checked by reading, not compiled, not run on hardware
//
#include <hip/hip_runtime.h>
#include <math.h>

typedef float     v2f  __attribute__((ext_vector_type(2)));
typedef float     v8f  __attribute__((ext_vector_type(8)));
typedef _Float16  v8h  __attribute__((ext_vector_type(8)));
typedef _Float16  v16h __attribute__((ext_vector_type(16)));

#define BATCH 8
#define QL 2048
#define KVL 2048
#define QD 512
#define KVD 512

#define EB_STRIDE  2052                      // f32 scores row stride (bank-spread)
#define PHF_STRIDE 2056                      // f16 probs row stride (4112B: 16B-aligned, banks m*4%64 distinct)
#define EB_BYTES   (16 * EB_STRIDE * 4)      // 131,328
#define PHF_BYTES  (16 * PHF_STRIDE * 2)     // 65,792
#define SMEM_BYTES (EB_BYTES + PHF_BYTES)    // 197,120  ( < 320KB WGP LDS )

// ---------------------------------------------------------------------------
// Kernel 0: kvhT[b][d][k] = (f16) kv[b][k][d]   (LDS 32x32 tile transpose)
// Grid: BATCH * (KVL/32) * (KVD/32) = 8192 blocks of 256 threads.
// ---------------------------------------------------------------------------
__global__ __launch_bounds__(256) void kv_to_half_T(const float* __restrict__ kv,
                                                    _Float16* __restrict__ kvhT) {
    __shared__ float t[32][33];
    const int blk = blockIdx.x;
    const int b   = blk >> 10;                 // / (64*16)
    const int kt  = (blk >> 4) & 63;           // k-tile
    const int dt  = blk & 15;                  // d-tile
    const int k0  = kt << 5, d0 = dt << 5;
    const int r   = threadIdx.x >> 5;          // 0..7
    const int c   = threadIdx.x & 31;

#pragma unroll
    for (int i = 0; i < 4; ++i) {
        const int row = r + 8 * i;
        t[row][c] = kv[((size_t)b * KVL + k0 + row) * KVD + d0 + c];   // coalesced
    }
    __syncthreads();
#pragma unroll
    for (int i = 0; i < 4; ++i) {
        const int row = r + 8 * i;             // d within tile
        kvhT[((size_t)b * KVD + d0 + row) * KVL + k0 + c] = (_Float16)t[c][row];
    }
}

// ---------------------------------------------------------------------------
// Kernel 1: q_proj[b,q,n] = sum_d q[b,q,d] * Wa[n,d]   (fp32 WMMA)
// ---------------------------------------------------------------------------
__global__ __launch_bounds__(256) void qproj_gemm(const float* __restrict__ q,
                                                  const float* __restrict__ Wa,
                                                  float* __restrict__ qp) {
    const int blk  = blockIdx.x;
    const int b    = blk >> 7;
    const int m0   = (blk & 127) << 4;
    const int lane = threadIdx.x & 31;
    const int wave = threadIdx.x >> 5;
    const int lo   = lane & 15;
    const int hi   = lane >> 4;
    const int koff = hi << 1;

    v8f acc[4] = {v8f{}, v8f{}, v8f{}, v8f{}};
    const float* qrow = q + ((size_t)b * QL + m0 + lo) * QD;

    for (int k0 = 0; k0 < QD; k0 += 4) {
        v2f a = *(const v2f*)(qrow + k0 + koff);
#pragma unroll
        for (int t = 0; t < 4; ++t) {
            const int n0 = wave * 64 + t * 16;
            v2f bf = *(const v2f*)(Wa + (size_t)(n0 + lo) * QD + k0 + koff);
            acc[t] = __builtin_amdgcn_wmma_f32_16x16x4_f32(
                false, a, false, bf, (short)0, acc[t], false, false);
        }
    }
#pragma unroll
    for (int t = 0; t < 4; ++t) {
        const int n = wave * 64 + t * 16 + lo;
#pragma unroll
        for (int r = 0; r < 8; ++r)
            qp[((size_t)b * QL + m0 + r + hi * 8) * KVD + n] = acc[t][r];
    }
}

// ---------------------------------------------------------------------------
// Kernel 2: per (b, 16-row q-tile):
//   e = q_proj @ kv^T  (fp32 WMMA -> LDS),  softmax -> attn + f16 probs LDS,
//   ctx = P(f16) @ kv(f16)  (f16 WMMA, f32 accum, B from transposed kvhT).
// ---------------------------------------------------------------------------
__global__ __launch_bounds__(256) void attn_kernel(const float* __restrict__ qp,
                                                   const float* __restrict__ kv,
                                                   const _Float16* __restrict__ kvhT,
                                                   const unsigned char* __restrict__ mask,
                                                   float* __restrict__ ctx,
                                                   float* __restrict__ attn) {
    extern __shared__ float smem[];
    float*    eb  = smem;                                   // [16][EB_STRIDE] f32 scores
    _Float16* phf = (_Float16*)(smem + 16 * EB_STRIDE);     // [16][PHF_STRIDE] f16 probs

    const int blk  = blockIdx.x;
    const int b    = blk >> 7;
    const int m0   = (blk & 127) << 4;
    const int lane = threadIdx.x & 31;
    const int wave = threadIdx.x >> 5;
    const int lo   = lane & 15;
    const int hi   = lane >> 4;
    const int koff = hi << 1;

    // ---- Phase A: scores (fp32 WMMA). Wave w owns e cols [256w, 256w+256) ----
    const float* qprow = qp + ((size_t)b * QL + m0 + lo) * KVD;
    for (int chunk = 0; chunk < 4; ++chunk) {
        v8f acc[4] = {v8f{}, v8f{}, v8f{}, v8f{}};
        for (int k0 = 0; k0 < KVD; k0 += 4) {
            v2f a = *(const v2f*)(qprow + k0 + koff);
#pragma unroll
            for (int t = 0; t < 4; ++t) {
                const int j0 = wave * 256 + chunk * 64 + t * 16;
                v2f bf = *(const v2f*)(kv + ((size_t)b * KVL + j0 + lo) * KVD + k0 + koff);
                acc[t] = __builtin_amdgcn_wmma_f32_16x16x4_f32(
                    false, a, false, bf, (short)0, acc[t], false, false);
            }
        }
#pragma unroll
        for (int t = 0; t < 4; ++t) {
            const int j = wave * 256 + chunk * 64 + t * 16 + lo;
#pragma unroll
            for (int r = 0; r < 8; ++r)
                eb[(r + hi * 8) * EB_STRIDE + j] = acc[t][r];
        }
    }
    __syncthreads();

    // ---- Phase B: softmax. Wave w handles rows 2w, 2w+1 ----
#pragma unroll
    for (int rr = 0; rr < 2; ++rr) {
        const int row  = wave * 2 + rr;
        const int qidx = m0 + row;
        const unsigned char* mrow = mask + ((size_t)b * QL + qidx) * KVL;
        const float* erow = eb + row * EB_STRIDE;
        _Float16*    prow = phf + row * PHF_STRIDE;

        float mx = -INFINITY;
        for (int j = lane; j < KVL; j += 32) {
            float v = mrow[j] ? -INFINITY : erow[j];
            mx = fmaxf(mx, v);
        }
#pragma unroll
        for (int s = 16; s; s >>= 1) mx = fmaxf(mx, __shfl_xor(mx, s, 32));

        float sum = 0.f;
        for (int j = lane; j < KVL; j += 32) {
            float v = mrow[j] ? -INFINITY : erow[j];
            sum += __expf(v - mx);
        }
#pragma unroll
        for (int s = 16; s; s >>= 1) sum += __shfl_xor(sum, s, 32);

        const float inv = 1.f / sum;
        float* arow = attn + ((size_t)b * QL + qidx) * KVL;
        for (int j = lane; j < KVL; j += 32) {
            float v = mrow[j] ? -INFINITY : erow[j];
            float p = __expf(v - mx) * inv;
            arow[j] = p;                 // fp32 attn output (coalesced)
            prow[j] = (_Float16)p;       // f16 probs for the P·V WMMA
        }
    }
    __syncthreads();

    // ---- Phase C: ctx = P @ kv  (f16 WMMA, f32 accum). Wave w owns d [64w,64w+64) ----
    v8f cacc[4] = {v8f{}, v8f{}, v8f{}, v8f{}};
    const _Float16* prow = phf + lo * PHF_STRIDE;
    for (int k0 = 0; k0 < KVL; k0 += 32) {
        // A frag 16x32 f16: lanes 0-15 carry K {0..15}, lanes 16-31 K {16..31}
        v8h a_lo = *(const v8h*)(prow + k0 + 16 * hi);
        v8h a_hi = *(const v8h*)(prow + k0 + 16 * hi + 8);
        v16h a = __builtin_shufflevector(a_lo, a_hi,
                                         0, 1, 2, 3, 4, 5, 6, 7,
                                         8, 9, 10, 11, 12, 13, 14, 15);
#pragma unroll
        for (int t = 0; t < 4; ++t) {
            const int d0 = wave * 64 + t * 16;
            // B frag 32x16 f16: lane n needs K=k0+16*hi .. +15 for fixed d -> contiguous in kvhT
            v16h bf = *(const v16h*)(kvhT + ((size_t)b * KVD + d0 + lo) * KVL + k0 + 16 * hi);
            cacc[t] = __builtin_amdgcn_wmma_f32_16x16x32_f16(
                false, a, false, bf, (short)0, cacc[t], false, false);
        }
    }
#pragma unroll
    for (int t = 0; t < 4; ++t) {
        const int d = wave * 64 + t * 16 + lo;
#pragma unroll
        for (int r = 0; r < 8; ++r)
            ctx[((size_t)b * QL + m0 + r + hi * 8) * QD + d] = cacc[t][r];
    }
}

// ---------------------------------------------------------------------------
extern "C" void kernel_launch(void* const* d_in, const int* in_sizes, int n_in,
                              void* d_out, int out_size, void* d_ws, size_t ws_size,
                              hipStream_t stream) {
    const float* q_seq  = (const float*)d_in[0];
    const float* kv_seq = (const float*)d_in[1];
    const float* Wa     = (const float*)d_in[2];
    const unsigned char* mask = (const unsigned char*)d_in[3];

    float* ctx  = (float*)d_out;                              // (B,QL,QD)
    float* attn = (float*)d_out + (size_t)BATCH * QL * QD;    // (B,QL,KVL)

    float*    qp   = (float*)d_ws;                                        // 33.5 MB
    _Float16* kvhT = (_Float16*)((float*)d_ws + (size_t)BATCH * QL * KVD); // 16.8 MB

    (void)in_sizes; (void)n_in; (void)out_size; (void)ws_size;

    hipFuncSetAttribute((const void*)attn_kernel,
                        hipFuncAttributeMaxDynamicSharedMemorySize, SMEM_BYTES);

    const int tblocks = BATCH * (KVL / 32) * (KVD / 32);   // 8192
    kv_to_half_T<<<tblocks, 256, 0, stream>>>(kv_seq, kvhT);

    const int blocks = BATCH * (QL / 16);                  // 1024
    qproj_gemm<<<blocks, 256, 0, stream>>>(q_seq, Wa, qp);
    attn_kernel<<<blocks, 256, SMEM_BYTES, stream>>>(qp, kv_seq, kvhT, mask, ctx, attn);
}
